// BTSTSSM_87600152969403
// MI455X (gfx1250) — matrix operand order repaired and checked by reading, not compiled, not run on hardware
//
#include <hip/hip_runtime.h>
#include <hip/hip_bf16.h>
#include <math.h>

typedef __attribute__((ext_vector_type(16))) __bf16 v16bf;
typedef __attribute__((ext_vector_type(8)))  float  v8f;

union BFrag {
  v16bf v;
  uint4 q[2];
};

__device__ __forceinline__ unsigned short f2bf(float f) {
  union { float f; unsigned u; } x; x.f = f;
  unsigned r = x.u + 0x7FFFu + ((x.u >> 16) & 1u);  // round-to-nearest-even
  return (unsigned short)(r >> 16);
}

__device__ __forceinline__ unsigned pk2bf(float a, float b) {
  return (unsigned)f2bf(a) | ((unsigned)f2bf(b) << 16);
}

__device__ __forceinline__ v8f wmma_bf16(const BFrag& a, const BFrag& b, v8f c) {
  // v_wmma_f32_16x16x32_bf16 : D = A(16x32) * B(32x16) + C
  return __builtin_amdgcn_wmma_f32_16x16x32_bf16(false, a.v, false, b.v,
                                                 (short)0, c, false, false);
}

// ---------------------------------------------------------------------------
// Coefficient precompute: A_bar / B_coeff (complex), indexed [hw][p]
// ---------------------------------------------------------------------------
__global__ void coef_kernel(const float* __restrict__ Lre, const float* __restrict__ Lim,
                            const float* __restrict__ values, const float* __restrict__ lstep,
                            float* __restrict__ Are, float* __restrict__ Aim,
                            float* __restrict__ Bre, float* __restrict__ Bim) {
  int idx = blockIdx.x * 256 + threadIdx.x;
  if (idx >= 32 * 32 * 64) return;
  int p = idx & 63;
  int hw = idx >> 6;
  int h = hw >> 5, w = hw & 31;
  float v0 = values[p * 4 + 0], v1 = values[p * 4 + 1];
  float v2 = values[p * 4 + 2], v3 = values[p * 4 + 3];
  float mx = fmaxf(fmaxf(v0, v1), fmaxf(v2, v3));
  float e0 = expf(v0 - mx), e1 = expf(v1 - mx), e2 = expf(v2 - mx), e3 = expf(v3 - mx);
  float inv = 4.0f / (e0 + e1 + e2 + e3);
  float xk = e0 * inv, yk = e1 * inv, zk = e2 * inv, wk = e3 * inv;
  float kb = (xk + yk - 2.f) * 0.25f;
  float kc = (xk + zk - 2.f) * 0.25f;
  float kd = (xk + wk - 2.f) * 0.125f;
  const float PI = 3.14159265358979323846f;
  float ch = 2.f * cosf(PI * (float)(h + 1) / 33.f);
  float cw = 2.f * cosf(PI * (float)(w + 1) / 33.f);
  float D = kb * cw + kc * ch + kd * ch * cw + 1.f;
  float lr = fminf(Lre[p], -1e-4f);
  float li = Lim[p];
  float st = expf(lstep[p]);
  float tr = lr * D, ti = li * D;
  float er = expf(tr * st);
  float ar = er * cosf(ti * st);
  float ai = er * sinf(ti * st);
  float rden = 1.f / (tr * tr + ti * ti);
  float br = ((ar - 1.f) * tr + ai * ti) * rden;
  float bi = (ai * tr - (ar - 1.f) * ti) * rden;
  Are[idx] = ar; Aim[idx] = ai; Bre[idx] = br; Bim[idx] = bi;
}

// ---------------------------------------------------------------------------
// DST matrices (bf16, row-major 32x32): Q, Qinv = conj(Q)^T, plus negated-imag
// ---------------------------------------------------------------------------
__global__ void qmat_kernel(unsigned short* QinvR, unsigned short* QinvI, unsigned short* QinvIN,
                            unsigned short* QR, unsigned short* QI, unsigned short* QIN) {
  int idx = blockIdx.x * 256 + threadIdx.x;
  if (idx >= 1024) return;
  int r = idx >> 5, c = idx & 31;  // Q[r][c]; k=r+1, j=c+1
  const float PI = 3.14159265358979323846f;
  float S = sinf(PI * (float)(r + 1) * (float)(c + 1) / 33.f) / sqrtf(16.5f);
  int ph = (r + 1) & 3;  // i^k
  float pr = (ph == 0) ? 1.f : (ph == 2) ? -1.f : 0.f;
  float pi = (ph == 1) ? 1.f : (ph == 3) ? -1.f : 0.f;
  float qre = pr * S, qim = pi * S;
  QR[idx] = f2bf(qre);
  QI[idx] = f2bf(qim);
  QIN[idx] = f2bf(-qim);
  QinvR[c * 32 + r] = f2bf(qre);     // Qinv[a][h] = conj(Q[h][a])
  QinvI[c * 32 + r] = f2bf(-qim);
  QinvIN[c * 32 + r] = f2bf(qim);
}

// ---------------------------------------------------------------------------
// Weight packing to bf16, K-transposed layout Wt[co][tap*Cin + ci]
// ---------------------------------------------------------------------------
__global__ void prep_wb(const float* __restrict__ Br, const float* __restrict__ Bi,
                        unsigned short* __restrict__ Wt) {
  int idx = blockIdx.x * 256 + threadIdx.x;
  if (idx >= 128 * 576) return;
  int co = idx / 576, k = idx % 576;
  int tap = k / 64, ci = k % 64;
  float v = (co < 64) ? Br[((size_t)tap * 64 + ci) * 64 + co]
                      : Bi[((size_t)tap * 64 + ci) * 64 + (co - 64)];
  Wt[idx] = f2bf(v);
}

__global__ void prep_wy(const float* __restrict__ Cr, const float* __restrict__ Ci,
                        const float* __restrict__ Dk, unsigned short* __restrict__ Wt) {
  int idx = blockIdx.x * 256 + threadIdx.x;
  if (idx >= 64 * 1728) return;
  int co = idx / 1728, k = idx % 1728;
  int tap = k / 192, cc = k % 192;
  float v;
  if (cc < 64)       v = 2.f  * Cr[((size_t)tap * 64 + cc) * 64 + co];
  else if (cc < 128) v = -2.f * Ci[((size_t)tap * 64 + (cc - 64)) * 64 + co];
  else               v =        Dk[((size_t)tap * 64 + (cc - 128)) * 64 + co];
  Wt[idx] = f2bf(v);
}

// ---------------------------------------------------------------------------
// Implicit-GEMM 3x3 SAME conv, bf16 WMMA, fp32 accumulate.
// Block = 128 thr (4 waves), tile = 2 pixel rows; wave = 16 pixels x COUT.
// MODE 0: in0 = u frames (256) / in1 = x0 frames (8), Cin=64, Cout=128 (re|im)
// MODE 1: in0 = xsp[...][128] / in1 = u (ch 128..191), Cin=192, Cout=64, gelu
// ---------------------------------------------------------------------------
template <int CIN, int COUT, int MODE>
__global__ __launch_bounds__(128) void conv_wmma(
    const float* __restrict__ in0, const float* __restrict__ in1,
    const unsigned short* __restrict__ Wt, float* __restrict__ out) {
  constexpr int K = 9 * CIN;
  constexpr int NT = COUT / 16;
  __shared__ __align__(16) unsigned short lact[4 * 34 * CIN];

  int tid = threadIdx.x;
  int img = blockIdx.x >> 4;
  int r0 = (blockIdx.x & 15) * 2;

  // Stage activation tile (4 rows incl. halo, 34 cols incl. halo) as bf16,
  // two channels per iteration -> one packed b32 LDS store, merged b64 load.
  constexpr int TOTAL2 = (4 * 34 * CIN) / 2;
  unsigned* lact32 = reinterpret_cast<unsigned*>(lact);
  for (int t2 = tid; t2 < TOTAL2; t2 += 128) {
    int t = t2 * 2;
    int c = t % CIN;
    int j = (t / CIN) % 34;
    int i = t / (CIN * 34);
    int gr = r0 - 1 + i;
    int gc = j - 1;
    float v0 = 0.f, v1 = 0.f;
    if (gr >= 0 && gr < 32 && gc >= 0 && gc < 32) {
      int hw = gr * 32 + gc;
      if (MODE == 0) {
        const float* src = (img < 256) ? &in0[((size_t)img * 1024 + hw) * 64 + c]
                                       : &in1[((size_t)(img - 256) * 1024 + hw) * 64 + c];
        v0 = src[0]; v1 = src[1];
      } else {
        const float* src = (c < 128) ? &in0[((size_t)img * 1024 + hw) * 128 + c]
                                     : &in1[((size_t)img * 1024 + hw) * 64 + (c - 128)];
        v0 = src[0]; v1 = src[1];
      }
    }
    lact32[t2] = pk2bf(v0, v1);
  }
  __syncthreads();

  int wave = tid >> 5, lane = tid & 31;
  int lr = wave >> 1, c0 = (wave & 1) * 16;
  int m = lane & 15, hi = lane >> 4;

  v8f z = {0.f, 0.f, 0.f, 0.f, 0.f, 0.f, 0.f, 0.f};
  v8f acc[NT];
#pragma unroll
  for (int n = 0; n < NT; n++) acc[n] = z;

  for (int kb = 0; kb < K / 32; kb++) {
    int kglob = kb * 32;
    int tap = kglob / CIN;
    int coff = kglob - tap * CIN;
    int kh = tap / 3, kw = tap - kh * 3;
    // A fragment (16x32 bf16): lane m = pixel, two contiguous 16B K-chunks
    BFrag a;
    const unsigned short* ap =
        &lact[(((lr + kh) * 34 + (c0 + m + kw)) * CIN) + coff];
    a.q[0] = *reinterpret_cast<const uint4*>(ap + hi * 8);
    a.q[1] = *reinterpret_cast<const uint4*>(ap + 16 + hi * 8);
#pragma unroll
    for (int nt = 0; nt < NT; nt++) {
      // B fragment (32x16 bf16): lane n = co, two 16B K-chunks at hi*8 and 16+hi*8
      BFrag b;
      const unsigned short* wp =
          Wt + (size_t)(nt * 16 + m) * K + kglob + hi * 8;
      b.q[0] = *reinterpret_cast<const uint4*>(wp);
      b.q[1] = *reinterpret_cast<const uint4*>(wp + 16);
      acc[nt] = wmma_bf16(a, b, acc[nt]);
    }
  }

  int row = r0 + lr;
#pragma unroll
  for (int nt = 0; nt < NT; nt++) {
#pragma unroll
    for (int j = 0; j < 8; j++) {
      int M = j + hi * 8;          // C/D layout: M = vgpr + 8*laneHi
      int col = c0 + M;
      float v = acc[nt][j];
      if (MODE == 1) {             // tanh-approx gelu
        float x3 = v * v * v;
        v = 0.5f * v * (1.f + tanhf(0.7978845608028654f * (v + 0.044715f * x3)));
      }
      out[((size_t)img * 1024 + row * 32 + col) * COUT + nt * 16 + m] = v;
    }
  }
}

// ---------------------------------------------------------------------------
// Complex 32x32 DST transform per (img,p) slice: X = Qm * S * Qm^T
// Block = 128 thr (4 waves), one slice per wave; 32 WMMAs per slice.
// in/out layout: [img][32*32][128] with re at ch p, im at ch 64+p.
// ---------------------------------------------------------------------------
__global__ __launch_bounds__(128) void dst_wmma(
    const float* __restrict__ in, float* __restrict__ out,
    const unsigned short* __restrict__ Qr, const unsigned short* __restrict__ Qi,
    const unsigned short* __restrict__ QiNeg) {
  __shared__ __align__(16) unsigned short lds[4][4096];
  int tid = threadIdx.x;
  int wave = tid >> 5, lane = tid & 31;
  int slice = blockIdx.x * 4 + wave;
  int img = slice >> 6;
  int p = slice & 63;
  unsigned short* L = lds[wave];  // [0..1023]=Sre^T [1024..]=Sim^T [2048..]=T1re [3072..]=T1im

  // Stage slice transposed ([col][row], contiguous K per column for B frags);
  // lane = column, two rows per iteration -> packed b32 LDS stores.
  {
    unsigned* L32r = reinterpret_cast<unsigned*>(L);
    unsigned* L32i = reinterpret_cast<unsigned*>(L + 1024);
    int c = lane;
    for (int r = 0; r < 32; r += 2) {
      size_t b0 = ((size_t)img * 1024 + (r * 32 + c)) * 128 + p;
      size_t b1 = b0 + 32 * 128;  // row r+1
      float re0 = in[b0],      re1 = in[b1];
      float im0 = in[b0 + 64], im1 = in[b1 + 64];
      L32r[(c * 32 + r) >> 1] = pk2bf(re0, re1);
      L32i[(c * 32 + r) >> 1] = pk2bf(im0, im1);
    }
  }
  __syncthreads();

  int m = lane & 15, hi = lane >> 4;
  v8f z = {0.f, 0.f, 0.f, 0.f, 0.f, 0.f, 0.f, 0.f};

  // Stage 1: T1 = Qm * S  (complex; -Qi*Si via pre-negated QiNeg as A)
#pragma unroll
  for (int mt = 0; mt < 2; mt++) {
#pragma unroll
    for (int nt = 0; nt < 2; nt++) {
      BFrag Ar, Ai, An, Brf, Bif;
      const unsigned short* qr = Qr + (mt * 16 + m) * 32;
      const unsigned short* qi = Qi + (mt * 16 + m) * 32;
      const unsigned short* qn = QiNeg + (mt * 16 + m) * 32;
      Ar.q[0] = *reinterpret_cast<const uint4*>(qr + hi * 8);
      Ar.q[1] = *reinterpret_cast<const uint4*>(qr + 16 + hi * 8);
      Ai.q[0] = *reinterpret_cast<const uint4*>(qi + hi * 8);
      Ai.q[1] = *reinterpret_cast<const uint4*>(qi + 16 + hi * 8);
      An.q[0] = *reinterpret_cast<const uint4*>(qn + hi * 8);
      An.q[1] = *reinterpret_cast<const uint4*>(qn + 16 + hi * 8);
      const unsigned short* br = &L[(nt * 16 + m) * 32 + hi * 8];
      const unsigned short* bi = &L[1024 + (nt * 16 + m) * 32 + hi * 8];
      Brf.q[0] = *reinterpret_cast<const uint4*>(br);
      Brf.q[1] = *reinterpret_cast<const uint4*>(br + 16);
      Bif.q[0] = *reinterpret_cast<const uint4*>(bi);
      Bif.q[1] = *reinterpret_cast<const uint4*>(bi + 16);
      v8f cr = wmma_bf16(Ar, Brf, z);
      cr = wmma_bf16(An, Bif, cr);   // - Qi * Si
      v8f ci = wmma_bf16(Ar, Bif, z);
      ci = wmma_bf16(Ai, Brf, ci);
#pragma unroll
      for (int j = 0; j < 8; j++) {
        int a = mt * 16 + j + hi * 8;
        int w = nt * 16 + m;
        L[2048 + a * 32 + w] = f2bf(cr[j]);  // row-major: A operand next stage
        L[3072 + a * 32 + w] = f2bf(ci[j]);
      }
    }
  }
  __syncthreads();

  // Stage 2: X = T1 * Qm^T  (B[k][n] = Qm[n][k] -> read Qm rows)
#pragma unroll
  for (int mt = 0; mt < 2; mt++) {
#pragma unroll
    for (int nt = 0; nt < 2; nt++) {
      BFrag Ar, Ai, Brf, Bif, Bn;
      const unsigned short* ar = &L[2048 + (mt * 16 + m) * 32];
      const unsigned short* ai = &L[3072 + (mt * 16 + m) * 32];
      Ar.q[0] = *reinterpret_cast<const uint4*>(ar + hi * 8);
      Ar.q[1] = *reinterpret_cast<const uint4*>(ar + 16 + hi * 8);
      Ai.q[0] = *reinterpret_cast<const uint4*>(ai + hi * 8);
      Ai.q[1] = *reinterpret_cast<const uint4*>(ai + 16 + hi * 8);
      const unsigned short* br = Qr + (nt * 16 + m) * 32 + hi * 8;
      const unsigned short* bi = Qi + (nt * 16 + m) * 32 + hi * 8;
      const unsigned short* bn = QiNeg + (nt * 16 + m) * 32 + hi * 8;
      Brf.q[0] = *reinterpret_cast<const uint4*>(br);
      Brf.q[1] = *reinterpret_cast<const uint4*>(br + 16);
      Bif.q[0] = *reinterpret_cast<const uint4*>(bi);
      Bif.q[1] = *reinterpret_cast<const uint4*>(bi + 16);
      Bn.q[0] = *reinterpret_cast<const uint4*>(bn);
      Bn.q[1] = *reinterpret_cast<const uint4*>(bn + 16);
      v8f cr = wmma_bf16(Ar, Brf, z);
      cr = wmma_bf16(Ai, Bn, cr);    // - T1i * Qi
      v8f ci = wmma_bf16(Ar, Bif, z);
      ci = wmma_bf16(Ai, Brf, ci);
#pragma unroll
      for (int j = 0; j < 8; j++) {
        int a = mt * 16 + j + hi * 8;
        int c = nt * 16 + m;
        size_t base = ((size_t)img * 1024 + a * 32 + c) * 128 + p;
        out[base] = cr[j];
        out[base + 64] = ci[j];
      }
    }
  }
}

// ---------------------------------------------------------------------------
// Sequential spectral-domain recurrence over L=32 (elementwise complex).
// x_0 = A*(Bc*x0h) + Bc*Bu_0 ; x_l = A*x_{l-1} + Bc*Bu_l
// ---------------------------------------------------------------------------
__global__ void scan_kernel(const float* __restrict__ bhat,
                            const float* __restrict__ Are, const float* __restrict__ Aim,
                            const float* __restrict__ Bre, const float* __restrict__ Bim,
                            float* __restrict__ xs) {
  int idx = blockIdx.x * 256 + threadIdx.x;  // 8*1024*64 = 524288
  int p = idx & 63;
  int hw = (idx >> 6) & 1023;
  int b = idx >> 16;
  int cidx = hw * 64 + p;
  float ar = Are[cidx], ai = Aim[cidx], br = Bre[cidx], bi = Bim[cidx];
  size_t x0off = ((size_t)(256 + b) * 1024 + hw) * 128 + p;
  float x0r = bhat[x0off], x0i = bhat[x0off + 64];
  float t0r = br * x0r - bi * x0i;
  float t0i = br * x0i + bi * x0r;
  float xr = ar * t0r - ai * t0i;
  float xi = ar * t0i + ai * t0r;
  {
    size_t off = ((size_t)b * 1024 + hw) * 128 + p;  // l=0 -> img = b
    float ur = bhat[off], ui = bhat[off + 64];
    xr += br * ur - bi * ui;
    xi += br * ui + bi * ur;
    xs[off] = xr; xs[off + 64] = xi;
  }
  for (int l = 1; l < 32; l++) {
    size_t off = ((size_t)(l * 8 + b) * 1024 + hw) * 128 + p;
    float ur = bhat[off], ui = bhat[off + 64];
    float bbr = br * ur - bi * ui;
    float bbi = br * ui + bi * ur;
    float nr = ar * xr - ai * xi + bbr;
    float ni = ar * xi + ai * xr + bbi;
    xr = nr; xi = ni;
    xs[off] = xr; xs[off + 64] = xi;
  }
}

// ---------------------------------------------------------------------------
extern "C" void kernel_launch(void* const* d_in, const int* in_sizes, int n_in,
                              void* d_out, int out_size, void* d_ws, size_t ws_size,
                              hipStream_t stream) {
  const float* u     = (const float*)d_in[0];
  const float* x0    = (const float*)d_in[1];
  const float* Lre   = (const float*)d_in[2];
  const float* Lim   = (const float*)d_in[3];
  const float* vals  = (const float*)d_in[4];
  const float* lstep = (const float*)d_in[5];
  const float* Br    = (const float*)d_in[6];
  const float* Bi    = (const float*)d_in[7];
  const float* Cr    = (const float*)d_in[8];
  const float* Ci    = (const float*)d_in[9];
  const float* Dk    = (const float*)d_in[10];

  char* ws = (char*)d_ws;
  float* Are = (float*)(ws + 0);                 // 4 x 65536 f32 coefs
  float* Aim = (float*)(ws + 262144);
  float* Bre = (float*)(ws + 524288);
  float* Bim = (float*)(ws + 786432);
  unsigned short* QinvR  = (unsigned short*)(ws + 1048576);  // 6 x 2KB DST mats
  unsigned short* QinvI  = (unsigned short*)(ws + 1048576 + 2048);
  unsigned short* QinvIN = (unsigned short*)(ws + 1048576 + 4096);
  unsigned short* QR     = (unsigned short*)(ws + 1048576 + 6144);
  unsigned short* QI     = (unsigned short*)(ws + 1048576 + 8192);
  unsigned short* QIN    = (unsigned short*)(ws + 1048576 + 10240);
  unsigned short* WtB    = (unsigned short*)(ws + 1060864);           // 128*576 bf16
  unsigned short* WtY    = (unsigned short*)(ws + 1060864 + 147456);  // 64*1728 bf16
  float* bufA = (float*)(ws + 1429504);                // 264*1024*128 f32
  float* bufB = (float*)(ws + 1429504 + 138412032);    // 264*1024*128 f32

  coef_kernel<<<256, 256, 0, stream>>>(Lre, Lim, vals, lstep, Are, Aim, Bre, Bim);
  qmat_kernel<<<4, 256, 0, stream>>>(QinvR, QinvI, QinvIN, QR, QI, QIN);
  prep_wb<<<(128 * 576 + 255) / 256, 256, 0, stream>>>(Br, Bi, WtB);
  prep_wy<<<(64 * 1728 + 255) / 256, 256, 0, stream>>>(Cr, Ci, Dk, WtY);

  // Bu = conv(u|x0, B_r) (+i) conv(u|x0, B_i): 264 imgs -> bufA[img][hw][128]
  conv_wmma<64, 128, 0><<<264 * 16, 128, 0, stream>>>(u, x0, WtB, bufA);
  // Bu_hat = Qinv * Bu * Qinv^T : 264*64 slices -> bufB
  dst_wmma<<<(264 * 64) / 4, 128, 0, stream>>>(bufA, bufB, QinvR, QinvI, QinvIN);
  // spectral recurrence -> bufA (imgs 0..255)
  scan_kernel<<<524288 / 256, 256, 0, stream>>>(bufB, Are, Aim, Bre, Bim, bufA);
  // x_sp = Q * xs * Q^T : 256*64 slices -> bufB
  dst_wmma<<<(256 * 64) / 4, 128, 0, stream>>>(bufA, bufB, QR, QI, QIN);
  // y = gelu(conv(xr,2Cr) + conv(xi,-2Ci) + conv(u,D)) -> d_out
  conv_wmma<192, 64, 1><<<256 * 16, 128, 0, stream>>>(bufB, u, WtY, (float*)d_out);
}
